// MaHS_88648124989690
// MI455X (gfx1250) — matrix operand order repaired and checked by reading, not compiled, not run on hardware
//
#include <hip/hip_runtime.h>

#define NB    16       // batch
#define HH    32
#define WW    32
#define NPX   1024     // H*W
#define TOK   16384    // B*NPX
#define CCH   64       // per-branch channels
#define DIMC  256
#define NHEAD 8
#define DHD   64
#define INNR  512

typedef __attribute__((ext_vector_type(16))) __bf16 v16bf;
typedef __attribute__((ext_vector_type(8)))  float  v8f;
typedef unsigned int u32x4 __attribute__((ext_vector_type(4)));
typedef int          i32x4 __attribute__((ext_vector_type(4)));
typedef int          i32x8 __attribute__((ext_vector_type(8)));

union U16 { v16bf v; __bf16 e[16]; };
union F8  { v8f   v; float  e[8];  };

#if defined(__has_builtin)
#if __has_builtin(__builtin_amdgcn_tensor_load_to_lds) && __has_builtin(__builtin_amdgcn_s_wait_tensorcnt)
#define HAVE_TDM 1
#endif
#endif

__device__ __forceinline__ float geluf(float v) {
    return 0.5f * v * (1.0f + erff(v * 0.70710678118654752f));
}
__device__ __forceinline__ float sigm(float v) {
    return 1.0f / (1.0f + __expf(-v));
}
__device__ __forceinline__ unsigned lds_off(const void* p) {
    return (unsigned)(unsigned long long)(__attribute__((address_space(3))) const void*)p;
}

// ---------------------------------------------------------------------------
// Weight conversion: fp32 -> bf16 (transposed where needed so GEMM sees [K,N])
// ---------------------------------------------------------------------------
__global__ void wconv_kernel(const float* wqkv, const float* wout,
                             const float* pw4, const float* ldw,
                             __bf16* o_qkv, __bf16* o_out, __bf16* o_pw4, __bf16* o_ldw)
{
    int i = blockIdx.x * blockDim.x + threadIdx.x;
    if (i < 98304) { o_qkv[i] = (__bf16)wqkv[i]; return; }
    i -= 98304;
    if (i < 32768) { o_out[i] = (__bf16)wout[i]; return; }
    i -= 32768;
    if (i < 4096) { int cin = i >> 6, cout = i & 63; o_pw4[i] = (__bf16)pw4[cout * 64 + cin]; return; }
    i -= 4096;
    if (i < 65536) { int cin = i >> 8, cout = i & 255; o_ldw[i] = (__bf16)ldw[cout * 256 + cin]; }
}

// ---------------------------------------------------------------------------
// Prep: channels-first LN (eps 1e-6) -> token-major fp32 xn, attention LN
// (eps 1e-5) -> bf16 a1, bf16 copy of branch-4 slice
// ---------------------------------------------------------------------------
__global__ void prep_kernel(const float* __restrict__ x,
                            const float* __restrict__ n1w, const float* __restrict__ n1b,
                            const float* __restrict__ alw, const float* __restrict__ alb,
                            float* __restrict__ xn, __bf16* __restrict__ a1, __bf16* __restrict__ x4b)
{
    int t = blockIdx.x * blockDim.x + threadIdx.x;
    if (t >= TOK) return;
    int b = t >> 10, p = t & 1023;
    const float* xp = x + (size_t)b * DIMC * NPX + p;
    float s = 0.f, s2 = 0.f;
    for (int c = 0; c < DIMC; ++c) { float v = xp[(size_t)c * NPX]; s += v; s2 += v * v; }
    float m = s / DIMC;
    float inv = rsqrtf(s2 / DIMC - m * m + 1e-6f);
    float* orow = xn + (size_t)t * DIMC;
    float a1s = 0.f, a1s2 = 0.f;
    for (int c = 0; c < DIMC; ++c) {
        float v = n1w[c] * ((xp[(size_t)c * NPX] - m) * inv) + n1b[c];
        orow[c] = v;
        if (c < CCH) { a1s += v; a1s2 += v * v; }
        if (c >= 192) x4b[(size_t)t * CCH + (c - 192)] = (__bf16)v;
    }
    float am = a1s / CCH;
    float ainv = rsqrtf(a1s2 / CCH - am * am + 1e-5f);
    for (int c = 0; c < CCH; ++c)
        a1[(size_t)t * CCH + c] = (__bf16)((orow[c] - am) * ainv * alw[c] + alb[c]);
}

// ---------------------------------------------------------------------------
// Generic WMMA GEMM: C[M,N] = A[M,K] * Wt[K,N] (+bias)(gelu)
// block 256 thr (8 waves), block tile 128x64, wave tile 16x64, K step 32.
// Wt tile staged in LDS (coalesced b128 loads); A read as two b128 per frag.
// STORE_MODE: 0 = f32 [M,N], 1 = bf16 [M,N], 2 = f32 NCHW (M = b*1024+p)
// ---------------------------------------------------------------------------
template<int STORE_MODE, bool HAS_BIAS, bool GELU_OUT>
__global__ __launch_bounds__(256) void gemm_wmma(
    const __bf16* __restrict__ A, const __bf16* __restrict__ Wt,
    const float* __restrict__ bias, float* __restrict__ Cf, __bf16* __restrict__ Cb,
    int K, int Nn)
{
    __shared__ __bf16 Wtile[32 * 64];
    int tid = threadIdx.x;
    int wv = tid >> 5, lane = tid & 31, hi = lane >> 4, lx = lane & 15;
    int rowb = blockIdx.x * 128 + wv * 16;
    int nb = blockIdx.y * 64;

    F8 acc[4];
#pragma unroll
    for (int nt = 0; nt < 4; ++nt)
#pragma unroll
        for (int j = 0; j < 8; ++j) acc[nt].e[j] = 0.f;

    int arow = rowb + lx;
    for (int k0 = 0; k0 < K; k0 += 32) {
        __syncthreads();
        {   // stage 32x64 W tile: one contiguous 16B load per thread
            int i = tid * 8;
            int r = i >> 6, cb = i & 63;
            const __bf16* src = Wt + (size_t)(k0 + r) * Nn + nb + cb;
#pragma unroll
            for (int j = 0; j < 8; ++j) Wtile[i + j] = src[j];
        }
        __syncthreads();
        U16 af;
        int ka = k0 + hi * 16;
        const __bf16* ap = A + (size_t)arow * K + ka;
#pragma unroll
        for (int h = 0; h < 8; ++h) { af.e[h] = ap[h]; af.e[h + 8] = ap[8 + h]; }
#pragma unroll
        for (int nt = 0; nt < 4; ++nt) {
            U16 bf_;
#pragma unroll
            for (int h = 0; h < 16; ++h) bf_.e[h] = Wtile[(hi * 16 + h) * 64 + nt * 16 + lx];
            acc[nt].v = __builtin_amdgcn_wmma_f32_16x16x32_bf16(
                false, af.v, false, bf_.v, (short)0, acc[nt].v, false, false);
        }
    }
#pragma unroll
    for (int nt = 0; nt < 4; ++nt) {
        int col = nb + nt * 16 + lx;
        float bvl = HAS_BIAS ? bias[col] : 0.f;
#pragma unroll
        for (int v = 0; v < 8; ++v) {
            int m = rowb + v + 8 * hi;
            float val = acc[nt].e[v] + bvl;
            if (GELU_OUT) val = geluf(val);
            if (STORE_MODE == 0)      Cf[(size_t)m * Nn + col] = val;
            else if (STORE_MODE == 1) Cb[(size_t)m * Nn + col] = (__bf16)val;
            else Cf[(size_t)(m >> 10) * Nn * NPX + (size_t)col * NPX + (m & 1023)] = val;
        }
    }
}

// ---------------------------------------------------------------------------
// Flash attention: one block per (b, h, 128-query chunk). 8 waves, each wave
// owns a 16-query tile; K/V staged 128 keys at a time in LDS via the Tensor
// Data Mover (D# built per ISA 8.3/8.4); online softmax; WMMA for QK^T & P*V.
// ---------------------------------------------------------------------------
__global__ __launch_bounds__(256) void attn_flash(const __bf16* __restrict__ qkv,
                                                  __bf16* __restrict__ attn_out)
{
    __shared__ __bf16 Qs[8192];
    __shared__ __bf16 Ks[8192];
    __shared__ __bf16 Vs[8192];
    __shared__ __bf16 Ps[8][512];

    int tid = threadIdx.x;
    int wv = tid >> 5, lane = tid & 31, hi = lane >> 4, lx = lane & 15;
    int qc = blockIdx.x & 7, h = (blockIdx.x >> 3) & 7, b = blockIdx.x >> 6;
    int q0 = qc * 128;
    const __bf16* base = qkv + (size_t)b * NPX * 1536;

    for (int i = tid; i < 8192; i += 256) {
        int r = i >> 6, d = i & 63;
        Qs[i] = base[(size_t)(q0 + r) * 1536 + h * 64 + d];
    }
    __syncthreads();

    U16 qa[2];
    int qrow = wv * 16 + lx;
#pragma unroll
    for (int f = 0; f < 2; ++f) {
        int ka = f * 32 + hi * 16;
#pragma unroll
        for (int h2 = 0; h2 < 8; ++h2) {
            qa[f].e[h2]     = Qs[qrow * 64 + ka + h2];
            qa[f].e[h2 + 8] = Qs[qrow * 64 + ka + 8 + h2];
        }
    }

    F8 acc[4];
#pragma unroll
    for (int nt = 0; nt < 4; ++nt)
#pragma unroll
        for (int j = 0; j < 8; ++j) acc[nt].e[j] = 0.f;
    float mrow[8], lrow[8];
#pragma unroll
    for (int v = 0; v < 8; ++v) { mrow[v] = -3.0e38f; lrow[v] = 0.f; }

    for (int ks = 0; ks < 8; ++ks) {
        __syncthreads();
#ifdef HAVE_TDM
        if (wv == 0) {
            // D# group1: data_size=2B, tensor 64 x 128 (stride 1536), tile 64x128
            i32x8 g1;
            g1[0] = 1 << 16;                 // workgroup_mask=0, data_size=1 (2B)
            g1[1] = 64 << 16;                // tensor_dim0 = 64 (bits 63:48)
            g1[2] = 128 << 16;               // tensor_dim0 hi=0 | tensor_dim1 = 128
            g1[3] = 64 << 16;                // tensor_dim1 hi=0 | tile_dim0 = 64
            g1[4] = 128;                     // tile_dim1 = 128, tile_dim2 = 0
            g1[5] = 1536;                    // tensor_dim0_stride low32
            g1[6] = 0;
            g1[7] = 0;
            i32x4 z4 = {};
            i32x8 z8 = {};
            unsigned long long gaK = (unsigned long long)(const void*)
                (base + ((size_t)(ks * 128) * 1536 + 512 + h * 64));
            unsigned long long gaV = gaK + 1024;  // +512 halves
            u32x4 g0;
            g0[0] = 1u;                                      // count=1
            g0[1] = lds_off(Ks);                             // lds_addr
            g0[2] = (unsigned)(gaK & 0xFFFFFFFFull);         // global_addr[31:0]
            g0[3] = (unsigned)((gaK >> 32) & 0x1FFFFFFull) | (2u << 30); // addr hi | type=2
            __builtin_amdgcn_tensor_load_to_lds(g0, g1, z4, z4, z8, 0);
            g0[1] = lds_off(Vs);
            g0[2] = (unsigned)(gaV & 0xFFFFFFFFull);
            g0[3] = (unsigned)((gaV >> 32) & 0x1FFFFFFull) | (2u << 30);
            __builtin_amdgcn_tensor_load_to_lds(g0, g1, z4, z4, z8, 0);
            __builtin_amdgcn_s_wait_tensorcnt(0);
        }
        __syncthreads();
#else
        for (int i = tid; i < 8192; i += 256) {
            int r = i >> 6, d = i & 63;
            size_t row = (size_t)(ks * 128 + r) * 1536;
            Ks[i] = base[row + 512 + h * 64 + d];
            Vs[i] = base[row + 1024 + h * 64 + d];
        }
        __syncthreads();
#endif

        for (int sub = 0; sub < 4; ++sub) {
            int kb = sub * 32;
            F8 st[2];
#pragma unroll
            for (int ct = 0; ct < 2; ++ct)
#pragma unroll
                for (int j = 0; j < 8; ++j) st[ct].e[j] = 0.f;

#pragma unroll
            for (int ct = 0; ct < 2; ++ct) {
                int keyrow = kb + ct * 16 + lx;
#pragma unroll
                for (int f = 0; f < 2; ++f) {
                    U16 bf_;
                    int d0 = f * 32 + hi * 16;
#pragma unroll
                    for (int h2 = 0; h2 < 16; ++h2) bf_.e[h2] = Ks[keyrow * 64 + d0 + h2];
                    st[ct].v = __builtin_amdgcn_wmma_f32_16x16x32_bf16(
                        false, qa[f].v, false, bf_.v, (short)0, st[ct].v, false, false);
                }
            }
            // online softmax per row (row = v + 8*hi, spread over 16 lanes)
#pragma unroll
            for (int v = 0; v < 8; ++v) {
                float s0 = st[0].e[v] * 0.125f, s1 = st[1].e[v] * 0.125f;
                float mx = fmaxf(s0, s1);
                for (int msk = 1; msk < 16; msk <<= 1) mx = fmaxf(mx, __shfl_xor(mx, msk, 16));
                float mnew = fmaxf(mrow[v], mx);
                float alpha = __expf(mrow[v] - mnew);
                float p0 = __expf(s0 - mnew), p1 = __expf(s1 - mnew);
                float rs = p0 + p1;
                for (int msk = 1; msk < 16; msk <<= 1) rs += __shfl_xor(rs, msk, 16);
                lrow[v] = lrow[v] * alpha + rs;
                mrow[v] = mnew;
#pragma unroll
                for (int nt = 0; nt < 4; ++nt) acc[nt].e[v] *= alpha;
                int pr = v + 8 * hi;
                Ps[wv][pr * 32 + lx]      = (__bf16)p0;
                Ps[wv][pr * 32 + 16 + lx] = (__bf16)p1;
            }
            asm volatile("s_wait_dscnt 0" ::: "memory");
            U16 pf;
            {
                int ka2 = hi * 16;
#pragma unroll
                for (int h2 = 0; h2 < 8; ++h2) {
                    pf.e[h2]     = Ps[wv][lx * 32 + ka2 + h2];
                    pf.e[h2 + 8] = Ps[wv][lx * 32 + ka2 + 8 + h2];
                }
            }
#pragma unroll
            for (int nt = 0; nt < 4; ++nt) {
                U16 vf;
                int col = nt * 16 + lx;
                int kk0 = kb + hi * 16;
#pragma unroll
                for (int h2 = 0; h2 < 16; ++h2) vf.e[h2] = Vs[(kk0 + h2) * 64 + col];
                acc[nt].v = __builtin_amdgcn_wmma_f32_16x16x32_bf16(
                    false, pf.v, false, vf.v, (short)0, acc[nt].v, false, false);
            }
        }
    }
#pragma unroll
    for (int v = 0; v < 8; ++v) {
        float inv = 1.f / lrow[v];
        int qg = q0 + wv * 16 + v + 8 * hi;
        size_t orow = ((size_t)b * NPX + qg) * INNR + h * 64;
#pragma unroll
        for (int nt = 0; nt < 4; ++nt)
            attn_out[orow + nt * 16 + lx] = (__bf16)(acc[nt].e[v] * inv);
    }
}

// ---------------------------------------------------------------------------
// gate xy: t1 = gelu(dwconv3x3(resize 8x8 -> 32x32) + b); g1 = pw(t1)
// ---------------------------------------------------------------------------
__device__ __forceinline__ float rz2d(const float* p, int c, int y, int x) {
    float sy = y * (7.f / 31.f), sx = x * (7.f / 31.f);
    int y0 = (int)floorf(sy); if (y0 > 6) y0 = 6; if (y0 < 0) y0 = 0;
    int x0 = (int)floorf(sx); if (x0 > 6) x0 = 6; if (x0 < 0) x0 = 0;
    float fy = sy - y0, fx = sx - x0;
    const float* pc = p + c * 64;
    float a = pc[y0 * 8 + x0], b2 = pc[y0 * 8 + x0 + 1];
    float c2 = pc[(y0 + 1) * 8 + x0], d2 = pc[(y0 + 1) * 8 + x0 + 1];
    return (a * (1 - fx) + b2 * fx) * (1 - fy) + (c2 * (1 - fx) + d2 * fx) * fy;
}

__global__ void gxyA_kernel(const float* __restrict__ pxy, const float* __restrict__ dww,
                            const float* __restrict__ dwb, float* __restrict__ t1)
{
    int i = blockIdx.x * blockDim.x + threadIdx.x;
    if (i >= CCH * NPX) return;
    int c = i >> 10, px = i & 1023, y = px >> 5, x = px & 31;
    float v = dwb[c];
    for (int dy = -1; dy <= 1; ++dy)
        for (int dx = -1; dx <= 1; ++dx) {
            int ny = y + dy, nx = x + dx;
            if (ny < 0 || ny >= 32 || nx < 0 || nx >= 32) continue;
            v += rz2d(pxy, c, ny, nx) * dww[c * 9 + (dy + 1) * 3 + (dx + 1)];
        }
    t1[i] = geluf(v);
}

__global__ void gxyB_kernel(const float* __restrict__ t1, const float* __restrict__ pww,
                            const float* __restrict__ pwb, float* __restrict__ g1)
{
    int i = blockIdx.x * blockDim.x + threadIdx.x;
    if (i >= CCH * NPX) return;
    int d = i >> 10, px = i & 1023;
    float v = pwb[d];
    for (int c = 0; c < CCH; ++c) v += t1[c * NPX + px] * pww[d * CCH + c];
    g1[i] = v;
}

// 1D gate (zx / zy): g[d][pos] = pw(gelu(dwconv1d(resize 8 -> 32)))
__global__ void gate1d_kernel(const float* __restrict__ params, const float* __restrict__ dww,
                              const float* __restrict__ dwb, const float* __restrict__ pww,
                              const float* __restrict__ pwb, float* __restrict__ gout)
{
    __shared__ float tsm[2048];
    int tid = threadIdx.x;
    for (int i = tid; i < 2048; i += blockDim.x) {
        int c = i >> 5, pos = i & 31;
        float v = dwb[c];
        for (int j = -1; j <= 1; ++j) {
            int np = pos + j;
            if (np < 0 || np >= 32) continue;
            float s = np * (7.f / 31.f);
            int i0 = (int)floorf(s); if (i0 > 6) i0 = 6; if (i0 < 0) i0 = 0;
            float f = s - i0;
            float rv = params[c * 8 + i0] * (1 - f) + params[c * 8 + i0 + 1] * f;
            v += rv * dww[c * 3 + j + 1];
        }
        tsm[i] = geluf(v);
    }
    __syncthreads();
    for (int i = tid; i < 2048; i += blockDim.x) {
        int d = i >> 5, pos = i & 31;
        float v = pwb[d];
        for (int c = 0; c < CCH; ++c) v += tsm[c * 32 + pos] * pww[d * CCH + c];
        gout[i] = v;
    }
}

// CBAM channel attention
__global__ void cbam_ca_kernel(const float* __restrict__ xn, const float* __restrict__ fc1,
                               const float* __restrict__ fc2, float* __restrict__ ca)
{
    __shared__ float pm[64], pmx[64], hid[8];
    int b = blockIdx.x, c = threadIdx.x;
    float s = 0.f, mx = -3.0e38f;
    for (int p = 0; p < NPX; ++p) {
        float v = xn[((size_t)(b << 10) + p) * DIMC + 64 + c];
        s += v; mx = fmaxf(mx, v);
    }
    pm[c] = s / NPX; pmx[c] = mx;
    __syncthreads();
    if (c < 8) {
        int j = c & 3;
        const float* src = (c < 4) ? pm : pmx;
        float hv = 0.f;
        for (int k = 0; k < CCH; ++k) hv += src[k] * fc1[j * CCH + k];
        hid[c] = fmaxf(hv, 0.f);
    }
    __syncthreads();
    float o = 0.f;
    for (int j = 0; j < 4; ++j) o += (hid[j] + hid[4 + j]) * fc2[c * 4 + j];
    ca[b * CCH + c] = sigm(o);
}

__global__ void spmap_kernel(const float* __restrict__ xn, const float* __restrict__ ca,
                             float* __restrict__ spm)
{
    int t = blockIdx.x * blockDim.x + threadIdx.x;
    if (t >= TOK) return;
    int b = t >> 10, p = t & 1023;
    float s = 0.f, mx = -3.0e38f;
    for (int c = 0; c < CCH; ++c) {
        float v = xn[(size_t)t * DIMC + 64 + c] * ca[b * CCH + c];
        s += v; mx = fmaxf(mx, v);
    }
    spm[b * 2048 + p] = s / CCH;
    spm[b * 2048 + 1024 + p] = mx;
}

__global__ void se_kernel(const float* __restrict__ xn, const float* __restrict__ fc1,
                          const float* __restrict__ fc2, float* __restrict__ sev)
{
    __shared__ float pm[64], hid[4];
    int b = blockIdx.x, c = threadIdx.x;
    float s = 0.f;
    for (int p = 0; p < NPX; ++p) s += xn[((size_t)(b << 10) + p) * DIMC + 128 + c];
    pm[c] = s / NPX;
    __syncthreads();
    if (c < 4) {
        float hv = 0.f;
        for (int k = 0; k < CCH; ++k) hv += pm[k] * fc1[c * CCH + k];
        hid[c] = fmaxf(hv, 0.f);
    }
    __syncthreads();
    float o = 0.f;
    for (int j = 0; j < 4; ++j) o += hid[j] * fc2[c * 4 + j];
    sev[b * CCH + c] = sigm(o);
}

__global__ void b4dw_kernel(const float* __restrict__ pw4, const float* __restrict__ dww,
                            const float* __restrict__ dwb, float* __restrict__ x4o)
{
    int i = blockIdx.x * blockDim.x + threadIdx.x;
    if (i >= TOK * CCH) return;
    int t = i >> 6, c = i & 63;
    int b = t >> 10, p = t & 1023, y = p >> 5, x = p & 31;
    float v = dwb[c];
    for (int dy = -1; dy <= 1; ++dy)
        for (int dx = -1; dx <= 1; ++dx) {
            int ny = y + dy, nx = x + dx;
            if (ny < 0 || ny >= 32 || nx < 0 || nx >= 32) continue;
            v += pw4[(((size_t)(b << 10) + ny * 32 + nx)) * CCH + c] * dww[c * 9 + (dy + 1) * 3 + (dx + 1)];
        }
    x4o[i] = v;
}

__global__ void combine_kernel(const float* __restrict__ xn, const float* __restrict__ aprj,
                               const float* __restrict__ g1, const float* __restrict__ ca,
                               const float* __restrict__ spm, const float* __restrict__ saw,
                               const float* __restrict__ gzx, const float* __restrict__ sev,
                               const float* __restrict__ gzy, const float* __restrict__ x4o,
                               float* __restrict__ cc)
{
    int t = blockIdx.x * blockDim.x + threadIdx.x;
    if (t >= TOK) return;
    int b = t >> 10, p = t & 1023, y = p >> 5, x = p & 31;
    float sa = 0.f;
    for (int ch = 0; ch < 2; ++ch)
        for (int dy = -3; dy <= 3; ++dy)
            for (int dx = -3; dx <= 3; ++dx) {
                int ny = y + dy, nx = x + dx;
                if (ny < 0 || ny >= 32 || nx < 0 || nx >= 32) continue;
                sa += spm[b * 2048 + ch * 1024 + ny * 32 + nx] * saw[ch * 49 + (dy + 3) * 7 + (dx + 3)];
            }
    float ssa = sigm(sa);
    float* orow = cc + (size_t)t * DIMC;
    const float* xrow = xn + (size_t)t * DIMC;
    for (int c = 0; c < CCH; ++c) {
        orow[c]       = aprj[(size_t)t * CCH + c] * g1[c * NPX + p];
        orow[64 + c]  = xrow[64 + c] * ca[b * CCH + c] * ssa * gzx[c * 32 + y];
        orow[128 + c] = xrow[128 + c] * sev[b * CCH + c] * gzy[c * 32 + x];
        orow[192 + c] = x4o[(size_t)t * CCH + c];
    }
}

__global__ void ln2_kernel(float* __restrict__ cc, const float* __restrict__ w,
                           const float* __restrict__ b)
{
    int t = blockIdx.x * blockDim.x + threadIdx.x;
    if (t >= TOK) return;
    float* row = cc + (size_t)t * DIMC;
    float s = 0.f, s2 = 0.f;
    for (int c = 0; c < DIMC; ++c) { float v = row[c]; s += v; s2 += v * v; }
    float m = s / DIMC;
    float inv = rsqrtf(s2 / DIMC - m * m + 1e-6f);
    for (int c = 0; c < DIMC; ++c) row[c] = w[c] * ((row[c] - m) * inv) + b[c];
}

__global__ void ldwconv_kernel(const float* __restrict__ cc, const float* __restrict__ dww,
                               const float* __restrict__ dwb, __bf16* __restrict__ z)
{
    int i = blockIdx.x * blockDim.x + threadIdx.x;
    if (i >= TOK * DIMC) return;
    int t = i >> 8, c = i & 255;
    int b = t >> 10, p = t & 1023, y = p >> 5, x = p & 31;
    float v = dwb[c];
    for (int dy = -1; dy <= 1; ++dy)
        for (int dx = -1; dx <= 1; ++dx) {
            int ny = y + dy, nx = x + dx;
            if (ny < 0 || ny >= 32 || nx < 0 || nx >= 32) continue;
            v += cc[(((size_t)(b << 10) + ny * 32 + nx)) * DIMC + c] * dww[c * 9 + (dy + 1) * 3 + (dx + 1)];
        }
    z[i] = (__bf16)geluf(v);
}

// ---------------------------------------------------------------------------
extern "C" void kernel_launch(void* const* d_in, const int* in_sizes, int n_in,
                              void* d_out, int out_size, void* d_ws, size_t ws_size,
                              hipStream_t stream)
{
    const float* x        = (const float*)d_in[0];
    const float* n1_w     = (const float*)d_in[1];
    const float* n1_b     = (const float*)d_in[2];
    const float* att_ln_w = (const float*)d_in[3];
    const float* att_ln_b = (const float*)d_in[4];
    const float* att_wqkv = (const float*)d_in[5];
    const float* att_wout = (const float*)d_in[6];
    const float* params_xy = (const float*)d_in[7];
    const float* cxy_dw_w = (const float*)d_in[8];
    const float* cxy_dw_b = (const float*)d_in[9];
    const float* cxy_pw_w = (const float*)d_in[10];
    const float* cxy_pw_b = (const float*)d_in[11];
    const float* ca_fc1   = (const float*)d_in[12];
    const float* ca_fc2   = (const float*)d_in[13];
    const float* sa_w     = (const float*)d_in[14];
    const float* params_zx = (const float*)d_in[15];
    const float* czx_dw_w = (const float*)d_in[16];
    const float* czx_dw_b = (const float*)d_in[17];
    const float* czx_pw_w = (const float*)d_in[18];
    const float* czx_pw_b = (const float*)d_in[19];
    const float* se_fc1   = (const float*)d_in[20];
    const float* se_fc2   = (const float*)d_in[21];
    const float* params_zy = (const float*)d_in[22];
    const float* czy_dw_w = (const float*)d_in[23];
    const float* czy_dw_b = (const float*)d_in[24];
    const float* czy_pw_w = (const float*)d_in[25];
    const float* czy_pw_b = (const float*)d_in[26];
    const float* dw_pw_w  = (const float*)d_in[27];
    const float* dw_pw_b  = (const float*)d_in[28];
    const float* dw_dw_w  = (const float*)d_in[29];
    const float* dw_dw_b  = (const float*)d_in[30];
    const float* n2_w     = (const float*)d_in[31];
    const float* n2_b     = (const float*)d_in[32];
    const float* ldw_dw_w = (const float*)d_in[33];
    const float* ldw_dw_b = (const float*)d_in[34];
    const float* ldw_pw_w = (const float*)d_in[35];
    const float* ldw_pw_b = (const float*)d_in[36];
    float* out = (float*)d_out;

    char* ws = (char*)d_ws;
    size_t off = 0;
    auto alloc = [&](size_t bytes) { char* p = ws + off; off += (bytes + 255) & ~(size_t)255; return p; };
    float*  xn     = (float*) alloc((size_t)TOK * DIMC * 4);
    __bf16* a1     = (__bf16*)alloc((size_t)TOK * CCH * 2);
    __bf16* x4b    = (__bf16*)alloc((size_t)TOK * CCH * 2);
    __bf16* wqkvB  = (__bf16*)alloc((size_t)64 * 1536 * 2);
    __bf16* woutB  = (__bf16*)alloc((size_t)512 * 64 * 2);
    __bf16* wpw4B  = (__bf16*)alloc((size_t)64 * 64 * 2);
    __bf16* wldwB  = (__bf16*)alloc((size_t)256 * 256 * 2);
    __bf16* qkv    = (__bf16*)alloc((size_t)TOK * 1536 * 2);
    __bf16* attn   = (__bf16*)alloc((size_t)TOK * INNR * 2);
    float*  aprj   = (float*) alloc((size_t)TOK * CCH * 4);
    float*  t1     = (float*) alloc((size_t)CCH * NPX * 4);
    float*  g1     = (float*) alloc((size_t)CCH * NPX * 4);
    float*  gzx    = (float*) alloc((size_t)CCH * 32 * 4);
    float*  gzy    = (float*) alloc((size_t)CCH * 32 * 4);
    float*  ca     = (float*) alloc((size_t)NB * CCH * 4);
    float*  sev    = (float*) alloc((size_t)NB * CCH * 4);
    float*  spm    = (float*) alloc((size_t)NB * 2 * NPX * 4);
    float*  pw4    = (float*) alloc((size_t)TOK * CCH * 4);
    float*  x4o    = (float*) alloc((size_t)TOK * CCH * 4);
    float*  cc     = (float*) alloc((size_t)TOK * DIMC * 4);
    __bf16* zb     = (__bf16*)alloc((size_t)TOK * DIMC * 2);
    (void)ws_size; (void)n_in; (void)in_sizes; (void)out_size;

    wconv_kernel<<<(200704 + 255) / 256, 256, 0, stream>>>(
        att_wqkv, att_wout, dw_pw_w, ldw_pw_w, wqkvB, woutB, wpw4B, wldwB);
    prep_kernel<<<64, 256, 0, stream>>>(x, n1_w, n1_b, att_ln_w, att_ln_b, xn, a1, x4b);
    // QKV projection (WMMA): [16384,64] x [64,1536] -> bf16
    gemm_wmma<1, false, false><<<dim3(128, 24), 256, 0, stream>>>(a1, wqkvB, nullptr, nullptr, qkv, 64, 1536);
    // flash attention (WMMA QK^T and P*V, TDM-staged K/V)
    attn_flash<<<1024, 256, 0, stream>>>(qkv, attn);
    // output projection: [16384,512] x [512,64] -> f32
    gemm_wmma<0, false, false><<<dim3(128, 1), 256, 0, stream>>>(attn, woutB, nullptr, aprj, nullptr, 512, 64);
    // learned-grid gates
    gxyA_kernel<<<256, 256, 0, stream>>>(params_xy, cxy_dw_w, cxy_dw_b, t1);
    gxyB_kernel<<<256, 256, 0, stream>>>(t1, cxy_pw_w, cxy_pw_b, g1);
    gate1d_kernel<<<1, 256, 0, stream>>>(params_zx, czx_dw_w, czx_dw_b, czx_pw_w, czx_pw_b, gzx);
    gate1d_kernel<<<1, 256, 0, stream>>>(params_zy, czy_dw_w, czy_dw_b, czy_pw_w, czy_pw_b, gzy);
    // CBAM / SE coefficients
    cbam_ca_kernel<<<NB, 64, 0, stream>>>(xn, ca_fc1, ca_fc2, ca);
    spmap_kernel<<<64, 256, 0, stream>>>(xn, ca, spm);
    se_kernel<<<NB, 64, 0, stream>>>(xn, se_fc1, se_fc2, sev);
    // branch 4: pointwise (WMMA, +bias+gelu) then depthwise 3x3
    gemm_wmma<0, true, true><<<dim3(128, 1), 256, 0, stream>>>(x4b, wpw4B, dw_pw_b, pw4, nullptr, 64, 64);
    b4dw_kernel<<<4096, 256, 0, stream>>>(pw4, dw_dw_w, dw_dw_b, x4o);
    // combine + LN2
    combine_kernel<<<64, 256, 0, stream>>>(xn, aprj, g1, ca, spm, sa_w, gzx, sev, gzy, x4o, cc);
    ln2_kernel<<<64, 256, 0, stream>>>(cc, n2_w, n2_b);
    // final depthwise 3x3 + gelu -> bf16, then final pointwise (WMMA) -> NCHW f32
    ldwconv_kernel<<<16384, 256, 0, stream>>>(cc, ldw_dw_w, ldw_dw_b, zb);
    gemm_wmma<2, true, false><<<dim3(128, 4), 256, 0, stream>>>(zb, wldwB, ldw_pw_b, out, nullptr, 256, 256);
}
